// MovieNetwork_41214506172690
// MI455X (gfx1250) — matrix operand order repaired and checked by reading, not compiled, not run on hardware
//
#include <hip/hip_runtime.h>

// ---------------------------------------------------------------------------
// Problem constants (from reference)
// ---------------------------------------------------------------------------
#define NN   4096      // N samples
#define LL   16        // name length
#define DD   768       // name embed dim
#define NHN  4         // name heads
#define DHN  192       // DD / NHN
#define HH   1024      // encoder hidden
#define NLAY 4         // encoder layers
#define CC   16        // classes
#define XF   256       // extra feats
#define NT   (NN * LL) // 65536 token rows

typedef __attribute__((ext_vector_type(16))) __bf16 v16bf;
typedef __attribute__((ext_vector_type(8)))  __bf16 v8bf;
typedef __attribute__((ext_vector_type(8)))  float  v8f;

// ---------------------------------------------------------------------------
// Generic WMMA GEMM:  C[M,N] = act(A[M,K] * W[N,K]^T + bias)
// A, W are bf16 row-major. Output optionally to f32 (Cf) and/or bf16 (Cb).
// Wave tile: (16*MT) x (16*NTI), compile-time, so the inner loop is a single
// straight-line block of loads + v_wmma with full EXEC (no divergent guards).
// Requires M % (16*MT) == 0, N % (16*NTI) == 0, K % 32 == 0.
// ---------------------------------------------------------------------------
template<int MT, int NTI>
__global__ __launch_bounds__(256)
void gemm_bf16_kernel(const __bf16* __restrict__ A, int lda,
                      const __bf16* __restrict__ W, int ldw,
                      const float* __restrict__ bias,
                      float* __restrict__ Cf, __bf16* __restrict__ Cb, int ldc,
                      int K, int relu, int total_tiles, int tiles_n)
{
    const int lane   = threadIdx.x & 31;
    const int wave   = (blockIdx.x * blockDim.x + threadIdx.x) >> 5;
    const int nwaves = (gridDim.x * blockDim.x) >> 5;
    const int r = lane & 15;       // row within A sub-tile / row of W (output col)
    const int g = lane >> 4;       // lane half

    for (int tile = wave; tile < total_tiles; tile += nwaves) {
        const int tm = tile / tiles_n;
        const int tn = tile - tm * tiles_n;
        const int m0 = tm * (16 * MT);
        const int n0 = tn * (16 * NTI);

        v8f acc[MT][NTI];
#pragma unroll
        for (int mi = 0; mi < MT; ++mi)
#pragma unroll
            for (int ni = 0; ni < NTI; ++ni)
                acc[mi][ni] = (v8f){0.f, 0.f, 0.f, 0.f, 0.f, 0.f, 0.f, 0.f};

        const __bf16* arow[MT];
#pragma unroll
        for (int mi = 0; mi < MT; ++mi)
            arow[mi] = A + (size_t)(m0 + 16 * mi + r) * lda;
        const __bf16* wbase = W + (size_t)(n0 + r) * ldw + 16 * g;

        for (int k0 = 0; k0 < K; k0 += 32) {
            // A fragment (16x32 bf16): lane half g covers K chunks
            // [16g, 16g+8) and [16g+8, 16g+16)
            v16bf af[MT];
#pragma unroll
            for (int mi = 0; mi < MT; ++mi) {
                v8bf alo = *(const v8bf*)(arow[mi] + k0 + 16 * g);
                v8bf ahi = *(const v8bf*)(arow[mi] + k0 + 16 * g + 8);
                af[mi] = __builtin_shufflevector(alo, ahi,
                          0, 1, 2, 3, 4, 5, 6, 7, 8, 9, 10, 11, 12, 13, 14, 15);
            }
#pragma unroll
            for (int ni = 0; ni < NTI; ++ni) {
                // B fragment (32x16 bf16): B[k,n] = W[n,k];
                // lane half g covers contiguous K [16g, 16g+16)
                v16bf bfv = *(const v16bf*)(wbase + (size_t)(16 * ni) * ldw + k0);
#pragma unroll
                for (int mi = 0; mi < MT; ++mi)
                    acc[mi][ni] = __builtin_amdgcn_wmma_f32_16x16x32_bf16(
                        false, af[mi], false, bfv, (short)0, acc[mi][ni],
                        false, false);
            }
        }

#pragma unroll
        for (int ni = 0; ni < NTI; ++ni) {
            const int col = n0 + 16 * ni + r;
            const float bv = bias ? bias[col] : 0.f;
#pragma unroll
            for (int mi = 0; mi < MT; ++mi) {
#pragma unroll
                for (int v = 0; v < 8; ++v) {
                    const int row = m0 + 16 * mi + v + 8 * g;
                    float val = acc[mi][ni][v] + bv;
                    if (relu) val = fmaxf(val, 0.f);
                    const size_t off = (size_t)row * ldc + col;
                    if (Cf) Cf[off] = val;
                    if (Cb) Cb[off] = (__bf16)val;
                }
            }
        }
    }
}

// ---------------------------------------------------------------------------
// Elementwise / gather kernels
// ---------------------------------------------------------------------------
__global__ __launch_bounds__(256)
void f2bf_kernel(const float* __restrict__ s, __bf16* __restrict__ d, long n)
{
    long i = (long)blockIdx.x * blockDim.x + threadIdx.x;
    const long stride = (long)gridDim.x * blockDim.x;
    for (; i < n; i += stride) d[i] = (__bf16)s[i];
}

__global__ __launch_bounds__(256)
void embed_kernel(const int* __restrict__ toks, const float* __restrict__ wte,
                  __bf16* __restrict__ emb, long total)
{
    long i = (long)blockIdx.x * blockDim.x + threadIdx.x;
    if (i >= total) return;
    const long row = i / DD;
    const int  d   = (int)(i - row * DD);
    emb[i] = (__bf16)wte[(size_t)toks[row] * DD + d];
}

// masked sum over sequence positions: name_emb[n,d] = sum_{l<len(n)} proj[n,l,d]
__global__ __launch_bounds__(256)
void masked_sum_kernel(const float* __restrict__ proj, const int* __restrict__ lens,
                       float* __restrict__ name_emb)
{
    const int idx = blockIdx.x * blockDim.x + threadIdx.x;
    if (idx >= NN * DD) return;
    const int n = idx / DD;
    const int d = idx - n * DD;
    const int len = lens[n];
    const float* p = proj + (size_t)n * LL * DD + d;
    float s = 0.f;
    for (int l = 0; l < len; ++l) s += p[(size_t)l * DD];
    name_emb[idx] = s;
}

// x[n, :256] = x_feats ; x[n, 256:1024] = name_emb   (as bf16)
__global__ __launch_bounds__(256)
void concat_kernel(const float* __restrict__ xf, const float* __restrict__ ne,
                   __bf16* __restrict__ x)
{
    const int idx = blockIdx.x * blockDim.x + threadIdx.x;
    if (idx >= NN * HH) return;
    const int n = idx / HH;
    const int c = idx - n * HH;
    const float v = (c < XF) ? xf[(size_t)n * XF + c]
                             : ne[(size_t)n * DD + (c - XF)];
    x[idx] = (__bf16)v;
}

// h = LayerNorm(h + delta) * g + b ; also emits bf16 copy
__global__ __launch_bounds__(256)
void ln_add_kernel(float* __restrict__ h, const float* __restrict__ delta,
                   const float* __restrict__ g, const float* __restrict__ b,
                   __bf16* __restrict__ hb)
{
    __shared__ float rs[8], rss[8];
    const int row = blockIdx.x;
    const int t   = threadIdx.x;
    float* hr = h + (size_t)row * HH;
    const float* dr = delta + (size_t)row * HH;

    float x[4], s = 0.f, ss = 0.f;
#pragma unroll
    for (int i = 0; i < 4; ++i) {
        const int c = t + 256 * i;
        const float v = hr[c] + dr[c];
        x[i] = v; s += v; ss += v * v;
    }
#pragma unroll
    for (int off = 16; off > 0; off >>= 1) {
        s  += __shfl_xor(s,  off, 32);
        ss += __shfl_xor(ss, off, 32);
    }
    if ((t & 31) == 0) { rs[t >> 5] = s; rss[t >> 5] = ss; }
    __syncthreads();
    float S = 0.f, SS = 0.f;
#pragma unroll
    for (int w = 0; w < 8; ++w) { S += rs[w]; SS += rss[w]; }
    const float mean = S * (1.f / HH);
    const float var  = SS * (1.f / HH) - mean * mean;
    const float inv  = rsqrtf(var + 1e-5f);

    __bf16* hbr = hb + (size_t)row * HH;
#pragma unroll
    for (int i = 0; i < 4; ++i) {
        const int c = t + 256 * i;
        const float v = (x[i] - mean) * inv * g[c] + b[c];
        hr[c] = v;
        hbr[c] = (__bf16)v;
    }
}

// ---------------------------------------------------------------------------
// Name attention: one block per (sample, head). S=16, dh=192, masked softmax.
// qkv is bf16 rows of width 2304 (q|k|v). Output o written bf16 (N*16, 768).
// ---------------------------------------------------------------------------
__global__ __launch_bounds__(256)
void attn_kernel(const __bf16* __restrict__ qkv, const int* __restrict__ lens,
                 __bf16* __restrict__ o)
{
    __shared__ float sq[LL][DHN], sk[LL][DHN], sv[LL][DHN];
    __shared__ float sa[LL][LL];

    const int blk = blockIdx.x;
    const int n  = blk >> 2;
    const int hh = blk & 3;
    const int t  = threadIdx.x;

    const __bf16* base = qkv + (size_t)n * LL * (3 * DD) + hh * DHN;
    for (int idx = t; idx < LL * DHN; idx += 256) {
        const int l = idx / DHN;
        const int d = idx - l * DHN;
        const __bf16* row = base + (size_t)l * (3 * DD) + d;
        sq[l][d] = (float)row[0];
        sk[l][d] = (float)row[DD];
        sv[l][d] = (float)row[2 * DD];
    }
    __syncthreads();

    const int len = lens[n];
    const int qi = t >> 4;
    const int ki = t & 15;
    float s = 0.f;
    for (int d = 0; d < DHN; ++d) s += sq[qi][d] * sk[ki][d];
    s *= 0.0721687836487032f;               // 1/sqrt(192)
    if (ki >= len) s = -1e9f;
    sa[qi][ki] = s;
    __syncthreads();

    float m = -3.4e38f;
#pragma unroll
    for (int j = 0; j < LL; ++j) m = fmaxf(m, sa[qi][j]);
    float sum = 0.f;
#pragma unroll
    for (int j = 0; j < LL; ++j) sum += __expf(sa[qi][j] - m);
    const float aval = __expf(s - m) / sum;
    __syncthreads();
    sa[qi][ki] = aval;
    __syncthreads();

    __bf16* orow = o + (size_t)n * LL * DD + hh * DHN;
    for (int idx = t; idx < LL * DHN; idx += 256) {
        const int l = idx / DHN;
        const int d = idx - l * DHN;
        float acc = 0.f;
#pragma unroll
        for (int k = 0; k < LL; ++k) acc += sa[l][k] * sv[k][d];
        orow[(size_t)l * DD + d] = (__bf16)acc;
    }
}

// ---------------------------------------------------------------------------
// Host launcher
// ---------------------------------------------------------------------------
static void launch_gemm(const __bf16* A, int lda, const __bf16* W, int ldw,
                        const float* bias, float* Cf, __bf16* Cb, int ldc,
                        int M, int N, int K, int relu, hipStream_t s)
{
    // All call sites: M % 32 == 0, N % 16 == 0, K % 32 == 0.
    const int tiles_m = M / 32;
    if ((N & 63) == 0) {
        const int tiles_n = N / 64;
        const int total   = tiles_m * tiles_n;
        int blocks = (total + 7) / 8;
        if (blocks > 16384) blocks = 16384;
        hipLaunchKernelGGL((gemm_bf16_kernel<2, 4>), dim3(blocks), dim3(256), 0, s,
                           A, lda, W, ldw, bias, Cf, Cb, ldc, K, relu, total, tiles_n);
    } else {
        const int tiles_n = N / 16;
        const int total   = tiles_m * tiles_n;
        int blocks = (total + 7) / 8;
        if (blocks > 16384) blocks = 16384;
        hipLaunchKernelGGL((gemm_bf16_kernel<2, 1>), dim3(blocks), dim3(256), 0, s,
                           A, lda, W, ldw, bias, Cf, Cb, ldc, K, relu, total, tiles_n);
    }
}

static void launch_f2bf(const float* src, __bf16* dst, long n, hipStream_t s)
{
    long blocks = (n + 255) / 256;
    if (blocks > 65535) blocks = 65535;
    hipLaunchKernelGGL(f2bf_kernel, dim3((unsigned)blocks), dim3(256), 0, s, src, dst, n);
}

extern "C" void kernel_launch(void* const* d_in, const int* in_sizes, int n_in,
                              void* d_out, int out_size, void* d_ws, size_t ws_size,
                              hipStream_t stream)
{
    const int*   name_tokens = (const int*)  d_in[0];
    const int*   name_lens   = (const int*)  d_in[1];
    const float* x_feats     = (const float*)d_in[2];
    const float* wte         = (const float*)d_in[3];
    const float* mha_in_w    = (const float*)d_in[4];
    const float* mha_in_b    = (const float*)d_in[5];
    const float* mha_out_w   = (const float*)d_in[6];
    const float* mha_out_b   = (const float*)d_in[7];
    const float* bott_w      = (const float*)d_in[8];
    const float* bott_b      = (const float*)d_in[9];
    const float* enc_in_w    = (const float*)d_in[10];
    const float* enc_in_b    = (const float*)d_in[11];
    const float* enc_out_w   = (const float*)d_in[12];
    const float* enc_out_b   = (const float*)d_in[13];
    const float* enc_ln1_g   = (const float*)d_in[14];
    const float* enc_ln1_b   = (const float*)d_in[15];
    const float* enc_ln2_g   = (const float*)d_in[16];
    const float* enc_ln2_b   = (const float*)d_in[17];
    const float* enc_ff1_w   = (const float*)d_in[18];
    const float* enc_ff1_b   = (const float*)d_in[19];
    const float* enc_ff2_w   = (const float*)d_in[20];
    const float* enc_ff2_b   = (const float*)d_in[21];
    const float* cls_w       = (const float*)d_in[22];
    const float* cls_b       = (const float*)d_in[23];
    float* out = (float*)d_out;

    // ---- workspace bump allocator (256B aligned) ----
    char* p = (char*)d_ws;
    auto alloc = [&](size_t bytes) -> char* {
        char* r = p;
        p += (bytes + 255) & ~(size_t)255;
        return r;
    };

    __bf16* emb_bf   = (__bf16*)alloc((size_t)NT * DD * 2);        // reused for attn output o
    char*   big      =          alloc((size_t)NT * 3 * DD * 2);    // qkv bf16, reused as proj f32
    __bf16* qkv_bf   = (__bf16*)big;
    float*  proj_f32 = (float*) big;
    float*  name_emb = (float*) alloc((size_t)NN * DD * 4);
    __bf16* x_bf     = (__bf16*)alloc((size_t)NN * HH * 2);
    float*  h_f32    = (float*) alloc((size_t)NN * HH * 4);
    __bf16* h_bf     = (__bf16*)alloc((size_t)NN * HH * 2);
    float*  tmp_f32  = (float*) alloc((size_t)NN * HH * 4);
    __bf16* tmp_bf   = (__bf16*)alloc((size_t)NN * 2 * HH * 2);    // max width 2048 (ff1)

    __bf16* w_mha_in  = (__bf16*)alloc((size_t)3 * DD * DD * 2);
    __bf16* w_mha_out = (__bf16*)alloc((size_t)DD * DD * 2);
    __bf16* w_bott    = (__bf16*)alloc((size_t)HH * HH * 2);
    __bf16* w_enc_in  = (__bf16*)alloc((size_t)NLAY * 3 * HH * HH * 2);
    __bf16* w_enc_out = (__bf16*)alloc((size_t)NLAY * HH * HH * 2);
    __bf16* w_ff1     = (__bf16*)alloc((size_t)NLAY * 2 * HH * HH * 2);
    __bf16* w_ff2     = (__bf16*)alloc((size_t)NLAY * HH * 2 * HH * 2);
    __bf16* w_cls     = (__bf16*)alloc((size_t)CC * HH * 2);

    // ---- 1. weight conversion f32 -> bf16 ----
    launch_f2bf(mha_in_w,  w_mha_in,  (long)3 * DD * DD,        stream);
    launch_f2bf(mha_out_w, w_mha_out, (long)DD * DD,            stream);
    launch_f2bf(bott_w,    w_bott,    (long)HH * HH,            stream);
    launch_f2bf(enc_in_w,  w_enc_in,  (long)NLAY * 3 * HH * HH, stream);
    launch_f2bf(enc_out_w, w_enc_out, (long)NLAY * HH * HH,     stream);
    launch_f2bf(enc_ff1_w, w_ff1,     (long)NLAY * 2 * HH * HH, stream);
    launch_f2bf(enc_ff2_w, w_ff2,     (long)NLAY * HH * 2 * HH, stream);
    launch_f2bf(cls_w,     w_cls,     (long)CC * HH,            stream);

    // ---- 2. embedding gather ----
    {
        const long total = (long)NT * DD;
        hipLaunchKernelGGL(embed_kernel, dim3((unsigned)((total + 255) / 256)), dim3(256),
                           0, stream, name_tokens, wte, emb_bf, total);
    }

    // ---- 3. QKV GEMM: (NT,768) x (2304,768)^T -> bf16 (NT,2304) ----
    launch_gemm(emb_bf, DD, w_mha_in, DD, mha_in_b,
                nullptr, qkv_bf, 3 * DD, NT, 3 * DD, DD, 0, stream);

    // ---- 4. masked attention per (sample, head); output overwrites emb buffer ----
    hipLaunchKernelGGL(attn_kernel, dim3(NN * NHN), dim3(256), 0, stream,
                       qkv_bf, name_lens, emb_bf);

    // ---- 5. out-projection: (NT,768) x (768,768)^T -> f32 (reuses qkv buffer) ----
    launch_gemm(emb_bf, DD, w_mha_out, DD, mha_out_b,
                proj_f32, nullptr, DD, NT, DD, DD, 0, stream);

    // ---- 6. masked sum over sequence ----
    hipLaunchKernelGGL(masked_sum_kernel, dim3((NN * DD + 255) / 256), dim3(256),
                       0, stream, proj_f32, name_lens, name_emb);

    // ---- 7. concat [x_feats | name_emb] -> bf16 (NN,1024) ----
    hipLaunchKernelGGL(concat_kernel, dim3((NN * HH + 255) / 256), dim3(256),
                       0, stream, x_feats, name_emb, x_bf);

    // ---- 8. bottleneck GEMM -> h (f32 + bf16) ----
    launch_gemm(x_bf, HH, w_bott, HH, bott_b,
                h_f32, h_bf, HH, NN, HH, HH, 0, stream);

    // ---- 9. encoder layers (seq len 1: MHA == out_w @ v_proj) ----
    for (int l = 0; l < NLAY; ++l) {
        const __bf16* wv  = w_enc_in  + ((size_t)l * 3 * HH + 2 * HH) * HH; // V slice
        const float*  bv  = enc_in_b  + (size_t)l * 3 * HH + 2 * HH;
        const __bf16* wo  = w_enc_out + (size_t)l * HH * HH;
        const float*  bo  = enc_out_b + (size_t)l * HH;
        const __bf16* w1  = w_ff1     + (size_t)l * 2 * HH * HH;
        const float*  b1  = enc_ff1_b + (size_t)l * 2 * HH;
        const __bf16* w2  = w_ff2     + (size_t)l * HH * 2 * HH;
        const float*  b2  = enc_ff2_b + (size_t)l * HH;

        // v = h @ Wv^T + bv   (bf16)
        launch_gemm(h_bf, HH, wv, HH, bv, nullptr, tmp_bf, HH, NN, HH, HH, 0, stream);
        // a = v @ Wo^T + bo   (f32)
        launch_gemm(tmp_bf, HH, wo, HH, bo, tmp_f32, nullptr, HH, NN, HH, HH, 0, stream);
        // h = LN(h + a)
        hipLaunchKernelGGL(ln_add_kernel, dim3(NN), dim3(256), 0, stream,
                           h_f32, tmp_f32,
                           enc_ln1_g + (size_t)l * HH, enc_ln1_b + (size_t)l * HH, h_bf);
        // f1 = relu(h @ W1^T + b1)  (bf16, width 2048)
        launch_gemm(h_bf, HH, w1, HH, b1, nullptr, tmp_bf, 2 * HH, NN, 2 * HH, HH, 1, stream);
        // f = f1 @ W2^T + b2  (f32)
        launch_gemm(tmp_bf, 2 * HH, w2, 2 * HH, b2, tmp_f32, nullptr, HH, NN, HH, 2 * HH, 0, stream);
        // h = LN(h + f)
        hipLaunchKernelGGL(ln_add_kernel, dim3(NN), dim3(256), 0, stream,
                           h_f32, tmp_f32,
                           enc_ln2_g + (size_t)l * HH, enc_ln2_b + (size_t)l * HH, h_bf);
    }

    // ---- 10. classifier: (NN,1024) x (16,1024)^T -> d_out f32 ----
    launch_gemm(h_bf, HH, w_cls, HH, cls_b, out, nullptr, CC, NN, CC, HH, 0, stream);

    (void)in_sizes; (void)n_in; (void)out_size; (void)ws_size;
}
